// SpUNetBase_80049600463357
// MI455X (gfx1250) — compile-verified, matrix operand order checked by reading, not run on hardware
//
#include <hip/hip_runtime.h>

// ============================================================================
// SpUNet forward for MI455X (gfx1250), wave32 + WMMA f16 (f32 accumulate).
//
// Roofline: ~440 GFLOP of gather-GEMM vs <50 MB resident data (fits in 192MB
// L2) -> matrix-pipe bound. All convolutions run through one gather-GEMM
// kernel built on v_wmma_f32_16x16x32_f16.
//
// Weights are pre-packed (once per conv, fused with f32->f16 convert) into the
// exact WMMA B-operand register image, so the GEMM inner loop is just:
//   2x global_load_b128 (A, gathered row)  +  4x global_load_b128 (B, packed,
//   coalesced)  +  2x v_wmma  -- no LDS staging, no barriers, no predication.
// Each wave produces two 16x16 output tiles from one A fetch.
//
// Param leaf order assumption (cannot be validated at runtime): top-level
// inputs in setup_inputs() dict order (d_in[0..12]); params flattened as a
// JAX pytree (dict keys sorted alphabetically, lists in order) -> 141 leaves
// at d_in[13+i]:
//   dec2 blk0: b1,b2,bp,g1,g2,gp,w1,w2,wp = 0..8 ; dec2 blk1: b1,b2,g1,g2,w1,w2 = 9..14
//   dec3 blk0: 15..23 ; dec3 blk1: 24..29
//   down[s]: b,g,w = 30+3s..32+3s (s=0..3)
//   enc: 15 blocks x (b1,b2,g1,g2,w1,w2) = 42..131 (stage bases 42,54,72,96)
//   stem_b=132, stem_g=133, stem_w=134
//   up2: b,g,w = 135..137 ; up3: b,g,w = 138..140
// ============================================================================

typedef __attribute__((ext_vector_type(16))) _Float16 v16h;
typedef __attribute__((ext_vector_type(8)))  _Float16 v8h;
typedef __attribute__((ext_vector_type(8)))  float    v8f;

#define BN_EPS 1e-3f

// --------------------------------------------------------------------------
// f32 -> f16 convert with zero-fill tail (activations / pad rows)
// --------------------------------------------------------------------------
__global__ void cvt_f16_kernel(const float* __restrict__ src, _Float16* __restrict__ dst,
                               long long n, long long ntot) {
  long long i = (long long)blockIdx.x * blockDim.x + threadIdx.x;
  if (i < ntot) dst[i] = (i < n) ? (_Float16)src[i] : (_Float16)0.f;
}

__global__ void zero_f32_kernel(float* __restrict__ p, long long n) {
  long long i = (long long)blockIdx.x * blockDim.x + threadIdx.x;
  if (i < n) p[i] = 0.f;
}

// --------------------------------------------------------------------------
// Pack f32 weights [K, Cin, Cout] into the WMMA B-operand register image
// (f16), zero-filling c >= Cin:
//   wS[(((k*nCh + cc)*nT + nt)*32 + lane)*16 + e]  =  B[kk][n]
//   with half=lane>>4, j=e>>1, p=e&1,
//        kk = ((j<4) ? 2j : 16+2(j-4)) + 8*half + p   (ISA 7.12.2 16-bit layout)
//        c  = cc*32 + kk,  n = nt*16 + (lane&15)
// In the GEMM, lane `l` then loads its 16 B halves as one contiguous 32-byte
// vector; consecutive lanes are consecutive 32B -> fully coalesced.
// --------------------------------------------------------------------------
__global__ void pack_b_kernel(const float* __restrict__ w, _Float16* __restrict__ wS,
                              int K, int Cin, int Cout, int nCh, int nT)
{
  long long i = (long long)blockIdx.x * blockDim.x + threadIdx.x;
  long long tot = (long long)K * nCh * nT * 512;
  if (i >= tot) return;
  int e    = (int)(i & 15);
  int lane = (int)((i >> 4) & 31);
  long long t = i >> 9;
  int nt = (int)(t % nT); t /= nT;
  int cc = (int)(t % nCh); t /= nCh;
  int k  = (int)t;
  int half = lane >> 4, j = e >> 1, p = e & 1;
  int kk = ((j < 4) ? 2 * j : 16 + 2 * (j - 4)) + 8 * half + p;
  int c  = cc * 32 + kk;
  int n  = nt * 16 + (lane & 15);
  wS[i] = (c < Cin) ? (_Float16)w[((long long)k * Cin + c) * Cout + n] : (_Float16)0.f;
}

// --------------------------------------------------------------------------
// Gather-GEMM via WMMA:  out[n,d] = sum_k sum_c feat16[idx[n,k], c] * w[k,c,d]
// feat16: [Nin+1, Cin] f16 (row Nin = zero pad row; tail tiles may read
//         arbitrary pad data, harmless: GEMM output rows are independent and
//         out-of-range rows are never stored).
// idx:    [Nout, K] int32, or nullptr for dense (row n gathers row n, K==1).
// wS:     packed B image (see pack_b_kernel).   out: [Nout, Cout] f32.
//
// Block = 4 waves (no LDS, no barriers); each wave owns one 16-row tile and
// TWO 16-col tiles (Cout % 32 == 0 for every layer).
// A per-lane (16-bit A 16x32): two contiguous 8-half runs
//   [c0+8*half .. +7] and [c0+16+8*half .. +7]  -> two global_load_b128
// (FULL path: Cin % 32 == 0, i.e. everything except the stem's Cin=6).
// C/D layout: VGPR v, lanes0-15 -> (M=v, N=lane); lanes16-31 -> (M=v+8).
// --------------------------------------------------------------------------
template <bool FULL>
__global__ __launch_bounds__(128) void gemm_gather_wmma(
    const _Float16* __restrict__ feat, const int* __restrict__ idx,
    const _Float16* __restrict__ wS, float* __restrict__ out,
    int Nout, int Nin, int Cin, int Cout, int K)
{
  const int lane = threadIdx.x & 31;
  const int wave = threadIdx.x >> 5;
  const int half = (lane >> 4) & 1;
  const int mrow = lane & 15;
  const int rowBase = (blockIdx.x * 4 + wave) * 16;
  const int n0 = blockIdx.y * 32;
  v8f acc0 = {0.f, 0.f, 0.f, 0.f, 0.f, 0.f, 0.f, 0.f};
  v8f acc1 = {0.f, 0.f, 0.f, 0.f, 0.f, 0.f, 0.f, 0.f};
  const int nCh = (Cin + 31) >> 5;
  const int nT  = Cout >> 4;
  const int r = rowBase + mrow;

  // per-lane base into the packed B image for tile pair (blockIdx.y*2, +1)
  const _Float16* bptr = wS + ((long long)(blockIdx.y * 2) * 32 + lane) * 16;
  const long long bstep = (long long)nT * 512;   // advance one (k,cc) chunk

  for (int k = 0; k < K; ++k) {
    long long g;
    if (idx) g = (r < Nout) ? (long long)idx[(long long)r * K + k] : (long long)Nin;
    else     g = (r < Nout) ? (long long)r : (long long)Nin;
    const _Float16* arow = feat + g * (long long)Cin;

    for (int cc = 0; cc < nCh; ++cc) {
      const int c0 = cc << 5;
      v16h a;
      if (FULL) {
        v8h alo = *(const v8h*)(arow + c0 + half * 8);
        v8h ahi = *(const v8h*)(arow + c0 + 16 + half * 8);
        a = __builtin_shufflevector(alo, ahi, 0, 1, 2, 3, 4, 5, 6, 7,
                                    8, 9, 10, 11, 12, 13, 14, 15);
      } else {                       // scalar fallback (stem, Cin = 6)
        #pragma unroll
        for (int j = 0; j < 8; ++j) {
          int kk = ((j < 4) ? (j * 2) : (16 + (j - 4) * 2)) + half * 8;
          int c = c0 + kk;
          a[2*j]   = (c     < Cin) ? arow[c]     : (_Float16)0.f;
          a[2*j+1] = (c + 1 < Cin) ? arow[c + 1] : (_Float16)0.f;
        }
      }
      v16h b0 = *(const v16h*)bptr;                 // 32B contiguous per lane
      v16h b1 = *(const v16h*)(bptr + 512);         // next 16-col tile
      bptr += bstep;
      acc0 = __builtin_amdgcn_wmma_f32_16x16x32_f16(false, a, false, b0,
                                                    (short)0, acc0, false, false);
      acc1 = __builtin_amdgcn_wmma_f32_16x16x32_f16(false, a, false, b1,
                                                    (short)0, acc1, false, false);
    }
  }
  #pragma unroll
  for (int v = 0; v < 8; ++v) {
    int rr = rowBase + half * 8 + v;
    if (rr < Nout) {
      out[(long long)rr * Cout + n0 + mrow]      = acc0[v];
      out[(long long)rr * Cout + n0 + 16 + mrow] = acc1[v];
    }
  }
}

// --------------------------------------------------------------------------
// BatchNorm (training mode) column stats: biased mean/var over N rows.
// grid.x = C/32, block = (32,8): coalesced 32-column stripes.
// --------------------------------------------------------------------------
__global__ void bn_stats_kernel(const float* __restrict__ x, float* __restrict__ mean,
                                float* __restrict__ var, int N, int C)
{
  __shared__ float ssum[8][32];
  __shared__ float ssq[8][32];
  const int c = blockIdx.x * 32 + threadIdx.x;
  float s = 0.f, q = 0.f;
  for (int n = threadIdx.y; n < N; n += 8) {
    float v = x[(long long)n * C + c];
    s += v; q += v * v;
  }
  ssum[threadIdx.y][threadIdx.x] = s;
  ssq[threadIdx.y][threadIdx.x]  = q;
  __syncthreads();
  if (threadIdx.y == 0) {
    for (int j = 1; j < 8; ++j) { s += ssum[j][threadIdx.x]; q += ssq[j][threadIdx.x]; }
    float m = s / (float)N;
    mean[c] = m;
    var[c]  = q / (float)N - m * m;
  }
}

// --------------------------------------------------------------------------
// Fused BN apply + optional residual + optional ReLU.
// Writes f32 master (out32) and/or f16 gather copy (out16), both at column
// offset `col` with row stride `ld` (implements concat for free).
// --------------------------------------------------------------------------
__global__ void bn_apply_kernel(const float* __restrict__ x, const float* __restrict__ mean,
                                const float* __restrict__ var, const float* __restrict__ gam,
                                const float* __restrict__ bet, const float* __restrict__ resid,
                                float* __restrict__ out32, _Float16* __restrict__ out16,
                                int N, int C, int ld, int col, int relu)
{
  long long i = (long long)blockIdx.x * blockDim.x + threadIdx.x;
  if (i >= (long long)N * C) return;
  int n = (int)(i / C), c = (int)(i % C);
  float y = (x[i] - mean[c]) * rsqrtf(var[c] + BN_EPS) * gam[c] + bet[c];
  if (resid) y += resid[i];
  if (relu)  y = fmaxf(y, 0.f);
  long long o = (long long)n * ld + col + c;
  if (out32) out32[o] = y;
  if (out16) out16[o] = (_Float16)y;
}

// copy skip features into concat buffers (f32 master + f16 gather copy)
__global__ void concat_copy_kernel(const float* __restrict__ src, float* __restrict__ out32,
                                   _Float16* __restrict__ out16, int N, int Cs, int ld, int col)
{
  long long i = (long long)blockIdx.x * blockDim.x + threadIdx.x;
  if (i >= (long long)N * Cs) return;
  int n = (int)(i / Cs), c = (int)(i % Cs);
  float v = src[i];
  long long o = (long long)n * ld + col + c;
  out32[o] = v;
  out16[o] = (_Float16)v;
}

// --------------------------------------------------------------------------
// SparseInverseConv: out[idx[m,k], :] += feat[m, :] @ w[k, :, :]
// One block per (m,k) pair; feat row in LDS; float atomic scatter.
// --------------------------------------------------------------------------
__global__ void inv_conv_kernel(const float* __restrict__ feat, const int* __restrict__ idx,
                                const float* __restrict__ w, float* __restrict__ out,
                                int M, int Cin, int Cout, int K)
{
  __shared__ float frow[256];
  const int m = blockIdx.x / K, k = blockIdx.x % K;
  for (int c = threadIdx.x; c < Cin; c += blockDim.x)
    frow[c] = feat[(long long)m * Cin + c];
  __syncthreads();
  const int dest = idx[(long long)m * K + k];
  for (int d = threadIdx.x; d < Cout; d += blockDim.x) {
    float s = 0.f;
    const float* wp = w + (long long)k * Cin * Cout + d;
    for (int c = 0; c < Cin; ++c) s += frow[c] * wp[(long long)c * Cout];
    atomicAdd(&out[(long long)dest * Cout + d], s);
  }
}

// --------------------------------------------------------------------------
// Final: grouped max over 2 samples, transpose to [BSS, C, FPS], gather xyz.
// --------------------------------------------------------------------------
__global__ void finalize_kernel(const float* __restrict__ x, const int* __restrict__ gidx,
                                const int* __restrict__ fidx, const float* __restrict__ coord,
                                float* __restrict__ out, int C, int FPS, long long xyz_off)
{
  const int i = blockIdx.x;                 // 0 .. BSS*FPS-1
  const int g0 = gidx[2 * i], g1 = gidx[2 * i + 1];
  const int b = i / FPS, f = i % FPS;
  for (int c = threadIdx.x; c < C; c += blockDim.x) {
    float v = fmaxf(x[(long long)g0 * C + c], x[(long long)g1 * C + c]);
    out[((long long)b * C + c) * FPS + f] = v;
  }
  if (threadIdx.x < 3)
    out[xyz_off + (long long)i * 3 + threadIdx.x] =
        coord[(long long)fidx[i] * 3 + threadIdx.x];
}

// ============================================================================
// Host orchestration
// ============================================================================
extern "C" void kernel_launch(void* const* d_in, const int* in_sizes, int n_in,
                              void* d_out, int out_size, void* d_ws, size_t ws_size,
                              hipStream_t stream)
{
  (void)in_sizes; (void)n_in; (void)out_size; (void)ws_size;
  const int N0 = 40000, N1 = 20000, N2 = 10000, N3 = 5000, N4 = 2500;

  const float* feat  = (const float*)d_in[0];
  const float* coord = (const float*)d_in[1];
  const int* stemIdx = (const int*)d_in[2];
  const int* subm[4] = {(const int*)d_in[3], (const int*)d_in[4],
                        (const int*)d_in[5], (const int*)d_in[6]};
  const int* down[4] = {(const int*)d_in[7], (const int*)d_in[8],
                        (const int*)d_in[9], (const int*)d_in[10]};
  const int* gidx = (const int*)d_in[11];
  const int* fidx = (const int*)d_in[12];
  auto P = [&](int i) -> const float* { return (const float*)d_in[13 + i]; };

  // stream-ordered bump allocator over d_ws (checkpoint/reset for temporaries;
  // safe because all kernels run in-order on `stream`)
  char* wsb = (char*)d_ws;
  size_t off = 0;
  auto alloc = [&](size_t bytes) -> void* {
    void* p = wsb + off;
    off += (bytes + 255) & ~(size_t)255;
    return p;
  };

  auto cvt = [&](const float* s, _Float16* d, long long n, long long nt) {
    cvt_f16_kernel<<<(unsigned)((nt + 255) / 256), 256, 0, stream>>>(s, d, n, nt);
  };
  auto bn = [&](const float* raw, const float* gam, const float* bet, const float* resid,
                float* o32, _Float16* o16, int N, int C, int ld, int col, int relu) {
    float* mv = (float*)alloc((size_t)2 * C * sizeof(float));
    bn_stats_kernel<<<C / 32, dim3(32, 8), 0, stream>>>(raw, mv, mv + C, N, C);
    long long tot = (long long)N * C;
    bn_apply_kernel<<<(unsigned)((tot + 255) / 256), 256, 0, stream>>>(
        raw, mv, mv + C, gam, bet, resid, o32, o16, N, C, ld, col, relu);
  };
  // gather-conv + BN (+relu/resid); weights packed to WMMA B image inline
  auto conv_bn = [&](const _Float16* f16, const int* idx, int K,
                     const float* w, const float* gam, const float* bet,
                     int Nout, int Nin, int Cin, int Cout,
                     const float* resid, int relu,
                     float* o32, _Float16* o16, int ld, int col) {
    size_t ck = off;
    const int nCh = (Cin + 31) / 32, nT = Cout / 16;
    long long wsn = (long long)K * nCh * nT * 512;
    _Float16* wS = (_Float16*)alloc((size_t)wsn * 2);
    pack_b_kernel<<<(unsigned)((wsn + 255) / 256), 256, 0, stream>>>(
        w, wS, K, Cin, Cout, nCh, nT);
    float* raw = (float*)alloc((size_t)Nout * Cout * 4);
    dim3 grid((Nout + 63) / 64, Cout / 32);
    if ((Cin & 31) == 0)
      gemm_gather_wmma<true><<<grid, 128, 0, stream>>>(f16, idx, wS, raw,
                                                       Nout, Nin, Cin, Cout, K);
    else
      gemm_gather_wmma<false><<<grid, 128, 0, stream>>>(f16, idx, wS, raw,
                                                        Nout, Nin, Cin, Cout, K);
    bn(raw, gam, bet, resid, o32, o16, Nout, Cout, ld, col, relu);
    off = ck;
  };
  // BasicBlock: relu(bn(conv1)) -> bn(conv2) + (proj(x) or x) -> relu
  auto block = [&](const float* x32, const _Float16* x16, const int* idx, int N,
                   int Cin, int Cout,
                   const float* w1, const float* g1, const float* b1,
                   const float* w2, const float* g2, const float* b2,
                   const float* wp, const float* gp, const float* bp,
                   float* o32, _Float16* o16, int ld, int col) {
    size_t ck = off;
    _Float16* h16 = (_Float16*)alloc((size_t)(N + 1) * Cout * 2);
    conv_bn(x16, idx, 27, w1, g1, b1, N, N, Cin, Cout, nullptr, 1, nullptr, h16, Cout, 0);
    const float* r32 = x32;
    if (wp) {  // projection: dense GEMM (idx = nullptr, K = 1), BN, no relu
      float* pr = (float*)alloc((size_t)N * Cout * 4);
      conv_bn(x16, nullptr, 1, wp, gp, bp, N, N, Cin, Cout, nullptr, 0, pr, nullptr, Cout, 0);
      r32 = pr;
    }
    conv_bn(h16, idx, 27, w2, g2, b2, N, N, Cout, Cout, r32, 1, o32, o16, ld, col);
    off = ck;
  };

  // ---------------- stem: SubMConv k=125 (6->32) + BN + ReLU ----------------
  _Float16* feat16 = (_Float16*)alloc((size_t)(N0 + 1) * 6 * 2);
  cvt(feat, feat16, (long long)N0 * 6, (long long)(N0 + 1) * 6);  // zero pad row
  float*    x32 = (float*)alloc((size_t)N0 * 32 * 4);
  _Float16* x16 = (_Float16*)alloc((size_t)(N0 + 1) * 32 * 2);
  conv_bn(feat16, stemIdx, 125, P(134), P(133), P(132), N0, N0, 6, 32,
          nullptr, 1, x32, x16, 32, 0);

  // ---------------- encoder ----------------
  const int ch[4]      = {32, 64, 128, 256};
  const int Ns[5]      = {N0, N1, N2, N3, N4};
  const int lay[4]     = {2, 3, 4, 6};
  const int encBase[4] = {42, 54, 72, 96};
  float* skip32[4];
  int cin = 32;
  for (int s = 0; s < 4; ++s) {
    const int N = Ns[s + 1], C = ch[s];
    float*    d32 = (float*)alloc((size_t)N * C * 4);
    _Float16* d16 = (_Float16*)alloc((size_t)(N + 1) * C * 2);
    conv_bn(x16, down[s], 8, P(32 + 3 * s), P(31 + 3 * s), P(30 + 3 * s),
            N, Ns[s], cin, C, nullptr, 1, d32, d16, C, 0);
    float* a32 = d32; _Float16* a16 = d16;
    for (int l = 0; l < lay[s]; ++l) {
      float*    o32 = (float*)alloc((size_t)N * C * 4);
      _Float16* o16 = (_Float16*)alloc((size_t)(N + 1) * C * 2);
      const int pb = encBase[s] + 6 * l;   // b1,b2,g1,g2,w1,w2
      block(a32, a16, subm[s], N, C, C,
            P(pb + 4), P(pb + 2), P(pb + 0),
            P(pb + 5), P(pb + 3), P(pb + 1),
            nullptr, nullptr, nullptr, o32, o16, C, 0);
      a32 = o32; a16 = o16;
    }
    skip32[s] = a32; x32 = a32; x16 = a16; cin = C;
  }

  // ---------------- decoder stage 3: N4 -> N3 ----------------
  {
    float* inv = (float*)alloc((size_t)N3 * 256 * 4);
    zero_f32_kernel<<<(unsigned)(((long long)N3 * 256 + 255) / 256), 256, 0, stream>>>(
        inv, (long long)N3 * 256);
    inv_conv_kernel<<<N4 * 8, 256, 0, stream>>>(x32, down[3], P(140), inv, N4, 256, 256, 8);
    const int CT = 384;                               // 256 + 128 concat
    float*    c32 = (float*)alloc((size_t)N3 * CT * 4);
    _Float16* c16 = (_Float16*)alloc((size_t)(N3 + 1) * CT * 2);
    { size_t ck = off; bn(inv, P(139), P(138), nullptr, c32, c16, N3, 256, CT, 0, 1); off = ck; }
    concat_copy_kernel<<<(unsigned)(((long long)N3 * 128 + 255) / 256), 256, 0, stream>>>(
        skip32[2], c32, c16, N3, 128, CT, 256);
    float*    b0_32 = (float*)alloc((size_t)N3 * 256 * 4);
    _Float16* b0_16 = (_Float16*)alloc((size_t)(N3 + 1) * 256 * 2);
    block(c32, c16, subm[2], N3, 384, 256,
          P(21), P(18), P(15), P(22), P(19), P(16), P(23), P(20), P(17),
          b0_32, b0_16, 256, 0);
    float*    b1_32 = (float*)alloc((size_t)N3 * 256 * 4);
    _Float16* b1_16 = (_Float16*)alloc((size_t)(N3 + 1) * 256 * 2);
    block(b0_32, b0_16, subm[2], N3, 256, 256,
          P(28), P(26), P(24), P(29), P(27), P(25), nullptr, nullptr, nullptr,
          b1_32, b1_16, 256, 0);
    x32 = b1_32; x16 = b1_16;
  }

  // ---------------- decoder stage 2: N3 -> N2 ----------------
  {
    float* inv = (float*)alloc((size_t)N2 * 288 * 4);
    zero_f32_kernel<<<(unsigned)(((long long)N2 * 288 + 255) / 256), 256, 0, stream>>>(
        inv, (long long)N2 * 288);
    inv_conv_kernel<<<N3 * 8, 256, 0, stream>>>(x32, down[2], P(137), inv, N3, 256, 288, 8);
    const int CT = 352;                               // 288 + 64 concat
    float*    c32 = (float*)alloc((size_t)N2 * CT * 4);
    _Float16* c16 = (_Float16*)alloc((size_t)(N2 + 1) * CT * 2);
    { size_t ck = off; bn(inv, P(136), P(135), nullptr, c32, c16, N2, 288, CT, 0, 1); off = ck; }
    concat_copy_kernel<<<(unsigned)(((long long)N2 * 64 + 255) / 256), 256, 0, stream>>>(
        skip32[1], c32, c16, N2, 64, CT, 288);
    float*    b0_32 = (float*)alloc((size_t)N2 * 288 * 4);
    _Float16* b0_16 = (_Float16*)alloc((size_t)(N2 + 1) * 288 * 2);
    block(c32, c16, subm[1], N2, 352, 288,
          P(6), P(3), P(0), P(7), P(4), P(1), P(8), P(5), P(2),
          b0_32, b0_16, 288, 0);
    float* b1_32 = (float*)alloc((size_t)N2 * 288 * 4);
    block(b0_32, b0_16, subm[1], N2, 288, 288,
          P(13), P(11), P(9), P(14), P(12), P(10), nullptr, nullptr, nullptr,
          b1_32, nullptr, 288, 0);
    x32 = b1_32;
  }

  // ---------------- ball-group max-pool + transpose + xyz gather ----------------
  finalize_kernel<<<4096, 288, 0, stream>>>(x32, gidx, fidx, coord, (float*)d_out,
                                            288, 1024, (long long)4 * 288 * 1024);
}
